// get_model_66666482369070
// MI455X (gfx1250) — compile-verified, matrix operand order checked by reading, not run on hardware
//
#include <hip/hip_runtime.h>
#include <hip/hip_bf16.h>

typedef __attribute__((ext_vector_type(16))) _Float16 v16h;
typedef __attribute__((ext_vector_type(8)))  _Float16 v8h;
typedef __attribute__((ext_vector_type(8)))  float    v8f;

#define BATCH 8
#define NPTS  2048

// ---------------------------------------------------------------------------
// Layer descriptor: Wt is W transposed + K-padded to f16, [Cout, Cin] row-major.
// Cin = padded input width (== X row stride, multiple of 32). Cout multiple of 16.
// ---------------------------------------------------------------------------
struct Layer {
  const _Float16* Wt;
  const float* b;
  const float* g;
  const float* bt;
  int Cin;
  int Cout;
};

// Load one WMMA A/B fragment (v16h) for v_wmma_f32_16x16x32_f16.
// Per ISA layout: lanes 0-15 hold K = koff..koff+7 and koff+16..koff+23 of row `row`;
// lanes 16-31 same row with koff shifted by +8 (caller folds hi*8 into koff).
__device__ __forceinline__ v16h load_frag(const _Float16* base, int row, int stride, int koff) {
  const _Float16* p = base + (size_t)row * stride + koff;
  v8h lo = *(const v8h*)(p);
  v8h hi = *(const v8h*)(p + 16);
  v16h r;
#pragma unroll
  for (int j = 0; j < 8; ++j) { r[j] = lo[j]; r[j + 8] = hi[j]; }
  return r;
}

// One MLP layer over K rows: Y = relu(g * ((X @ W + b) * inv) + bt), f16 in/out.
// 4 waves (128 threads) partition the (K/16)x(Cout/16) output tiles.
__device__ void layer_wmma(const _Float16* X, _Float16* Y, const Layer L, int K, int tid) {
  const int lane = tid & 31;
  const int wave = tid >> 5;
  const int l0   = lane & 15;
  const int hi8  = (lane >> 4) * 8;
  const int mt_n = K >> 4;
  const int nt_n = L.Cout >> 4;
  const float inv = 0.99999500003750f;  // 1/sqrt(1 + 1e-5)
  for (int t = wave; t < mt_n * nt_n; t += 4) {
    const int mt = t / nt_n;
    const int nt = t - mt * nt_n;
    v8f acc = {};
    for (int kk = 0; kk < L.Cin; kk += 32) {
      v16h a = load_frag(X,    mt * 16 + l0, L.Cin, kk + hi8);
      v16h w = load_frag(L.Wt, nt * 16 + l0, L.Cin, kk + hi8);
      acc = __builtin_amdgcn_wmma_f32_16x16x32_f16(false, a, false, w, (short)0, acc, false, false);
    }
    const int n = nt * 16 + l0;
    const float bb = L.b[n], gg = L.g[n], tt = L.bt[n];
#pragma unroll
    for (int r = 0; r < 8; ++r) {
      const int m = mt * 16 + r + hi8;
      float y = gg * ((acc[r] + bb) * inv) + tt;
      Y[(size_t)m * L.Cout + n] = (_Float16)(y > 0.f ? y : 0.f);
    }
  }
}

// ---------------------------------------------------------------------------
// Farthest point sampling: one block per batch, writes sampled coordinates.
// Matches reference scan semantics: first pick is index 0; each output is the
// pick BEFORE the distance update of that iteration.
// ---------------------------------------------------------------------------
__global__ void fps_kernel(const float* __restrict__ xyz, int M, int npoint,
                           float* __restrict__ out_xyz) {
  __shared__ float dist[2048];
  __shared__ float rv[256];
  __shared__ int   ri[256];
  __shared__ int   cur;
  const int b = blockIdx.x, tid = threadIdx.x;
  for (int i = tid; i < M; i += 256) dist[i] = 1e10f;
  if (tid == 0) cur = 0;
  __syncthreads();
  for (int it = 0; it < npoint; ++it) {
    const int far = cur;
    const float cx = xyz[((size_t)b * M + far) * 3 + 0];
    const float cy = xyz[((size_t)b * M + far) * 3 + 1];
    const float cz = xyz[((size_t)b * M + far) * 3 + 2];
    if (tid < 3) out_xyz[((size_t)b * npoint + it) * 3 + tid] = xyz[((size_t)b * M + far) * 3 + tid];
    float best = -1.f; int bi = M;
    for (int i = tid; i < M; i += 256) {
      const float dx = xyz[((size_t)b * M + i) * 3 + 0] - cx;
      const float dy = xyz[((size_t)b * M + i) * 3 + 1] - cy;
      const float dz = xyz[((size_t)b * M + i) * 3 + 2] - cz;
      const float nd = fminf(dist[i], dx * dx + dy * dy + dz * dz);
      dist[i] = nd;
      if (nd > best) { best = nd; bi = i; }  // strided i increasing -> first max per thread
    }
    rv[tid] = best; ri[tid] = bi;
    __syncthreads();
    for (int off = 128; off > 0; off >>= 1) {
      if (tid < off) {
        if (rv[tid + off] > rv[tid] || (rv[tid + off] == rv[tid] && ri[tid + off] < ri[tid])) {
          rv[tid] = rv[tid + off]; ri[tid] = ri[tid + off];
        }
      }
      __syncthreads();
    }
    if (tid == 0) cur = ri[0];
    __syncthreads();
  }
}

// ---------------------------------------------------------------------------
// Weight prep: f32 W[Cin,Cout] -> f16 Wt[Cout,CinPad] (transposed, zero-padded K).
// ---------------------------------------------------------------------------
__global__ void prep_wt(const float* __restrict__ W, _Float16* __restrict__ Wt,
                        int Cin, int Cout, int CinPad) {
  const int i = blockIdx.x * blockDim.x + threadIdx.x;
  if (i >= Cout * CinPad) return;
  const int n = i / CinPad, k = i - n * CinPad;
  Wt[i] = (k < Cin) ? (_Float16)W[(size_t)k * Cout + n] : (_Float16)0.f;
}

// ---------------------------------------------------------------------------
// Fused set-abstraction scale: one workgroup per (centroid s, batch b).
// ball-query (wave0 ballot compaction) -> gather (centered xyz [+feats]) into LDS
// -> 3 WMMA MLP layers ping-ponging in LDS -> max over K -> f32 features out.
// Dynamic LDS: [0,512) gidx, then bufA (aHalfs halfs), then bufB.
// ---------------------------------------------------------------------------
__global__ void sa_fused(const float* __restrict__ src_xyz, int Msrc,
                         const float* __restrict__ src_feat, int featC, int featStride,
                         const float* __restrict__ new_xyz, int S,
                         float r2, int K, int CinPad,
                         Layer L0, Layer L1, Layer L2,
                         float* __restrict__ out_pts, int outStride, int outOff,
                         int aHalfs) {
  extern __shared__ __align__(16) char smem[];
  int* gidx       = (int*)smem;
  _Float16* bufA  = (_Float16*)(smem + 512);
  _Float16* bufB  = bufA + aHalfs;
  const int s = blockIdx.x, b = blockIdx.y;
  const int tid = threadIdx.x;

  const float cx = new_xyz[((size_t)b * S + s) * 3 + 0];
  const float cy = new_xyz[((size_t)b * S + s) * 3 + 1];
  const float cz = new_xyz[((size_t)b * S + s) * 3 + 2];

  // --- ball query: first K in-ball indices in ascending index order (wave 0) ---
  if (tid < 32) {
    int cnt = 0;
    for (int base = 0; base < Msrc && cnt < K; base += 32) {
      const int i = base + tid;
      bool inball = false;
      if (i < Msrc) {
        const float dx = src_xyz[((size_t)b * Msrc + i) * 3 + 0] - cx;
        const float dy = src_xyz[((size_t)b * Msrc + i) * 3 + 1] - cy;
        const float dz = src_xyz[((size_t)b * Msrc + i) * 3 + 2] - cz;
        inball = (dx * dx + dy * dy + dz * dz) <= r2;
      }
      const unsigned m = (unsigned)__ballot(inball);
      const int pre = __popc(m & ((1u << tid) - 1u));
      if (inball && (cnt + pre) < K) gidx[cnt + pre] = i;
      cnt += __popc(m);
    }
    if (tid == 0) {  // pad with first in-ball index (reference semantics)
      int stored = cnt < K ? cnt : K;
      int first  = stored > 0 ? gidx[0] : 0;
      for (int j = stored; j < K; ++j) gidx[j] = first;
    }
  }
  __syncthreads();

  // --- gather grouped features into bufA [K, CinPad] f16, zero-padded ---
  for (int e = tid; e < K * CinPad; e += blockDim.x) {
    const int row = e / CinPad, c = e - row * CinPad;
    const int gi = gidx[row];
    float v = 0.f;
    if (c < featC) {
      v = src_feat[((size_t)b * Msrc + gi) * featStride + c];
    } else {
      const int cc = c - featC;
      if (cc < 3) {
        const float cen = (cc == 0) ? cx : ((cc == 1) ? cy : cz);
        v = src_xyz[((size_t)b * Msrc + gi) * 3 + cc] - cen;
      }
    }
    bufA[(size_t)row * CinPad + c] = (_Float16)v;
  }
  __syncthreads();

  // --- 3 MLP layers, LDS ping-pong ---
  layer_wmma(bufA, bufB, L0, K, tid);
  __syncthreads();
  layer_wmma(bufB, bufA, L1, K, tid);
  __syncthreads();
  layer_wmma(bufA, bufB, L2, K, tid);
  __syncthreads();

  // --- max pool over K, write f32 features (relu output >= 0) ---
  for (int c = tid; c < L2.Cout; c += blockDim.x) {
    float mx = (float)bufB[c];
    for (int r = 1; r < K; ++r) mx = fmaxf(mx, (float)bufB[(size_t)r * L2.Cout + c]);
    out_pts[((size_t)b * S + s) * outStride + outOff + c] = mx;
  }
}

// ---------------------------------------------------------------------------
// Generic global-memory WMMA GEMM + BN + ReLU (for SA3 / FC where M is small).
// One wave per 16x16 output tile. grid = (M/16, Cout/16).
// ---------------------------------------------------------------------------
__global__ void gemm_bn_relu(const _Float16* __restrict__ X, Layer L,
                             _Float16* __restrict__ Y) {
  const int tid = threadIdx.x;
  const int l0  = tid & 15;
  const int hi8 = (tid >> 4) * 8;
  const int mt = blockIdx.x, nt = blockIdx.y;
  __builtin_prefetch(L.Wt + (size_t)(nt * 16) * L.Cin, 0, 1);
  v8f acc = {};
  for (int kk = 0; kk < L.Cin; kk += 32) {
    v16h a = load_frag(X,    mt * 16 + l0, L.Cin, kk + hi8);
    v16h w = load_frag(L.Wt, nt * 16 + l0, L.Cin, kk + hi8);
    acc = __builtin_amdgcn_wmma_f32_16x16x32_f16(false, a, false, w, (short)0, acc, false, false);
  }
  const int n = nt * 16 + l0;
  const float inv = 0.99999500003750f;
  const float bb = L.b[n], gg = L.g[n], tt = L.bt[n];
#pragma unroll
  for (int r = 0; r < 8; ++r) {
    const int m = mt * 16 + r + hi8;
    float y = gg * ((acc[r] + bb) * inv) + tt;
    Y[(size_t)m * L.Cout + n] = (_Float16)(y > 0.f ? y : 0.f);
  }
}

// SA3 input: [1024 rows, 672] f16 = [l2_xyz(3) | l2_pts(640) | zeros(29)] per point.
__global__ void sa3_build(const float* __restrict__ l2_xyz, const float* __restrict__ l2_pts,
                          _Float16* __restrict__ X) {
  const int i = blockIdx.x * blockDim.x + threadIdx.x;
  if (i >= 1024 * 672) return;
  const int row = i / 672, c = i - row * 672;
  float v = 0.f;
  if (c < 3)        v = l2_xyz[row * 3 + c];
  else if (c < 643) v = l2_pts[(size_t)row * 640 + (c - 3)];
  X[i] = (_Float16)v;
}

// SA3 global max over 128 points -> l3 f32 (to d_out) + f16 fc input (rows 8..15 zero).
__global__ void sa3_pool(const _Float16* __restrict__ Y, float* __restrict__ l3,
                         _Float16* __restrict__ fcX) {
  const int i = blockIdx.x * blockDim.x + threadIdx.x;
  if (i >= 16 * 1024) return;
  const int b = i >> 10, c = i & 1023;
  if (b < 8) {
    float mx = (float)Y[((size_t)b * 128) * 1024 + c];
    for (int r = 1; r < 128; ++r) mx = fmaxf(mx, (float)Y[((size_t)(b * 128 + r)) * 1024 + c]);
    l3[b * 1024 + c] = mx;
    fcX[i] = (_Float16)mx;
  } else {
    fcX[i] = (_Float16)0.f;
  }
}

// Final linear 256 -> 40 (no BN/ReLU), f32 logits.
__global__ void fc3_kernel(const _Float16* __restrict__ X, const float* __restrict__ W,
                           const float* __restrict__ bias, float* __restrict__ out) {
  const int i = blockIdx.x * blockDim.x + threadIdx.x;
  if (i >= 8 * 40) return;
  const int b = i / 40, n = i - b * 40;
  float s = bias[n];
  for (int k = 0; k < 256; ++k) s += (float)X[b * 256 + k] * W[(size_t)k * 40 + n];
  out[i] = s;
}

// ---------------------------------------------------------------------------
// Host orchestration.
// Param pytree flattening assumed: nested dicts key-sorted ->
//   fc1(W,b,bt,g), fc2, fc3, sa1[3 branches][3 layers](W,b,bt,g), sa2[...], sa3[3].
// Top level: xyz first (insertion order); fall back to xyz-last if sizes say so.
// ---------------------------------------------------------------------------
extern "C" void kernel_launch(void* const* d_in, const int* in_sizes, int n_in,
                              void* d_out, int out_size, void* d_ws, size_t ws_size,
                              hipStream_t stream) {
  (void)out_size; (void)ws_size;
  int xi = 0, p0 = 1;
  if (n_in > 0 && in_sizes[0] != BATCH * NPTS * 3) { xi = n_in - 1; p0 = 0; }
  const float* xyz = (const float*)d_in[xi];

  char* ws = (char*)d_ws;
  size_t off = 0;
  auto alloc = [&](size_t bytes) -> void* {
    off = (off + 255) & ~(size_t)255;
    void* p = ws + off;
    off += bytes;
    return p;
  };

  float*     l1_xyz = (float*)alloc((size_t)8 * 512 * 3 * 4);
  float*     l2_xyz = (float*)alloc((size_t)8 * 128 * 3 * 4);
  float*     l1_pts = (float*)alloc((size_t)8 * 512 * 320 * 4);
  float*     l2_pts = (float*)alloc((size_t)8 * 128 * 640 * 4);
  _Float16*  sa3X   = (_Float16*)alloc((size_t)1024 * 672 * 2);
  _Float16*  y0     = (_Float16*)alloc((size_t)1024 * 256 * 2);
  _Float16*  y1     = (_Float16*)alloc((size_t)1024 * 512 * 2);
  _Float16*  y2     = (_Float16*)alloc((size_t)1024 * 1024 * 2);
  _Float16*  fcX    = (_Float16*)alloc((size_t)16 * 1024 * 2);
  _Float16*  fcY    = (_Float16*)alloc((size_t)16 * 512 * 2);
  _Float16*  fcZ    = (_Float16*)alloc((size_t)16 * 256 * 2);

  auto mk = [&](int leaf, int cinTrue, int cinPad, int cout) -> Layer {
    _Float16* Wt = (_Float16*)alloc((size_t)cout * cinPad * 2);
    const int total = cout * cinPad;
    prep_wt<<<dim3((total + 255) / 256), dim3(256), 0, stream>>>(
        (const float*)d_in[leaf], Wt, cinTrue, cout, cinPad);
    Layer L;
    L.Wt = Wt;
    L.b  = (const float*)d_in[leaf + 1];
    L.bt = (const float*)d_in[leaf + 2];
    L.g  = (const float*)d_in[leaf + 3];
    L.Cin = cinPad;
    L.Cout = cout;
    return L;
  };

  const int fc1L = p0 + 0, fc2L = p0 + 4, fc3L = p0 + 8;
  const int sa1L = p0 + 12, sa2L = p0 + 48, sa3L = p0 + 84;

  Layer s1[3][3] = {
      {mk(sa1L + 0, 3, 32, 32),   mk(sa1L + 4, 32, 32, 32),    mk(sa1L + 8, 32, 32, 64)},
      {mk(sa1L + 12, 3, 32, 64),  mk(sa1L + 16, 64, 64, 64),   mk(sa1L + 20, 64, 64, 128)},
      {mk(sa1L + 24, 3, 32, 64),  mk(sa1L + 28, 64, 64, 96),   mk(sa1L + 32, 96, 96, 128)},
  };
  Layer s2[3][3] = {
      {mk(sa2L + 0, 323, 352, 64),   mk(sa2L + 4, 64, 64, 64),      mk(sa2L + 8, 64, 64, 128)},
      {mk(sa2L + 12, 323, 352, 128), mk(sa2L + 16, 128, 128, 128),  mk(sa2L + 20, 128, 128, 256)},
      {mk(sa2L + 24, 323, 352, 128), mk(sa2L + 28, 128, 128, 128),  mk(sa2L + 32, 128, 128, 256)},
  };
  Layer s3[3] = {mk(sa3L + 0, 643, 672, 256), mk(sa3L + 4, 256, 256, 512),
                 mk(sa3L + 8, 512, 512, 1024)};
  Layer f1 = mk(fc1L, 1024, 1024, 512);
  Layer f2 = mk(fc2L, 512, 512, 256);

  // ---- SA1 ----
  fps_kernel<<<dim3(8), dim3(256), 0, stream>>>(xyz, 2048, 512, l1_xyz);
  {
    const int aH = 16 * 32, bH = 16 * 64;  // bufA: max(32,32); bufB: max(32,64)
    sa_fused<<<dim3(512, 8), dim3(128), 512 + (size_t)(aH + bH) * 2, stream>>>(
        xyz, 2048, xyz, 0, 0, l1_xyz, 512, 0.01f, 16, 32,
        s1[0][0], s1[0][1], s1[0][2], l1_pts, 320, 0, aH);
  }
  {
    const int aH = 32 * 64, bH = 32 * 128;
    sa_fused<<<dim3(512, 8), dim3(128), 512 + (size_t)(aH + bH) * 2, stream>>>(
        xyz, 2048, xyz, 0, 0, l1_xyz, 512, 0.04f, 32, 32,
        s1[1][0], s1[1][1], s1[1][2], l1_pts, 320, 64, aH);
  }
  {
    const int aH = 128 * 96, bH = 128 * 128;
    sa_fused<<<dim3(512, 8), dim3(128), 512 + (size_t)(aH + bH) * 2, stream>>>(
        xyz, 2048, xyz, 0, 0, l1_xyz, 512, 0.16f, 128, 32,
        s1[2][0], s1[2][1], s1[2][2], l1_pts, 320, 192, aH);
  }

  // ---- SA2 ----
  fps_kernel<<<dim3(8), dim3(256), 0, stream>>>(l1_xyz, 512, 128, l2_xyz);
  {
    const int aH = 32 * 352, bH = 32 * 128;
    sa_fused<<<dim3(128, 8), dim3(128), 512 + (size_t)(aH + bH) * 2, stream>>>(
        l1_xyz, 512, l1_pts, 320, 320, l2_xyz, 128, 0.04f, 32, 352,
        s2[0][0], s2[0][1], s2[0][2], l2_pts, 640, 0, aH);
  }
  {
    const int aH = 64 * 352, bH = 64 * 256;
    sa_fused<<<dim3(128, 8), dim3(128), 512 + (size_t)(aH + bH) * 2, stream>>>(
        l1_xyz, 512, l1_pts, 320, 320, l2_xyz, 128, 0.16f, 64, 352,
        s2[1][0], s2[1][1], s2[1][2], l2_pts, 640, 128, aH);
  }
  {
    const int aH = 128 * 352, bH = 128 * 256;  // ~152KB dynamic LDS (320KB WGP budget)
    sa_fused<<<dim3(128, 8), dim3(128), 512 + (size_t)(aH + bH) * 2, stream>>>(
        l1_xyz, 512, l1_pts, 320, 320, l2_xyz, 128, 0.64f, 128, 352,
        s2[2][0], s2[2][1], s2[2][2], l2_pts, 640, 384, aH);
  }

  // ---- SA3 (group_all) ----
  sa3_build<<<dim3((1024 * 672 + 255) / 256), dim3(256), 0, stream>>>(l2_xyz, l2_pts, sa3X);
  gemm_bn_relu<<<dim3(64, 16), dim3(32), 0, stream>>>(sa3X, s3[0], y0);
  gemm_bn_relu<<<dim3(64, 32), dim3(32), 0, stream>>>(y0, s3[1], y1);
  gemm_bn_relu<<<dim3(64, 64), dim3(32), 0, stream>>>(y1, s3[2], y2);
  sa3_pool<<<dim3((16 * 1024 + 255) / 256), dim3(256), 0, stream>>>(
      y2, (float*)d_out + 320, fcX);

  // ---- FC head ----
  gemm_bn_relu<<<dim3(1, 32), dim3(32), 0, stream>>>(fcX, f1, fcY);
  gemm_bn_relu<<<dim3(1, 16), dim3(32), 0, stream>>>(fcY, f2, fcZ);
  fc3_kernel<<<dim3(2), dim3(160), 0, stream>>>(
      fcZ, (const float*)d_in[fc3L], (const float*)d_in[fc3L + 1], (float*)d_out);
}
